// GATModel_75788992905467
// MI455X (gfx1250) — matrix operand order repaired and checked by reading, not compiled, not run on hardware
//
#include <hip/hip_runtime.h>
#include <hip/hip_bf16.h>

// ---------------- types for WMMA ----------------
typedef __attribute__((ext_vector_type(16))) _Float16 v16h;
typedef __attribute__((ext_vector_type(8)))  _Float16 v8h;
typedef __attribute__((ext_vector_type(8)))  float    v8f;

#define NEG_SLOPE 0.2f

// ---------------- small utility kernels ----------------
__global__ void fill_f32_kernel(float* __restrict__ p, float v, int n) {
    int i = blockIdx.x * blockDim.x + threadIdx.x;
    if (i < n) p[i] = v;
}

__global__ void fill_i32_kernel(int* __restrict__ p, int v, int n) {
    int i = blockIdx.x * blockDim.x + threadIdx.x;
    if (i < n) p[i] = v;
}

__global__ void cvt_f32_to_f16_kernel(const float* __restrict__ src,
                                      _Float16* __restrict__ dst, int n) {
    int i = blockIdx.x * blockDim.x + threadIdx.x;
    if (i < n) dst[i] = (_Float16)src[i];
}

// W is [K rows][Ncols cols] row-major f32 -> Wt is [Ncols][K] f16 (per-column contiguous K)
__global__ void transpose_cvt_kernel(const float* __restrict__ W,
                                     _Float16* __restrict__ Wt,
                                     int K, int Ncols) {
    int i = blockIdx.x * blockDim.x + threadIdx.x;
    if (i >= K * Ncols) return;
    int k = i / Ncols;
    int n = i % Ncols;
    Wt[(size_t)n * K + k] = (_Float16)W[i];
}

// ---------------- WMMA GEMM: C[M x 128] = A[M x K] * B[K x 128] ----------------
// One wave (block of 32) computes a full 16x128 output strip: the A fragment for
// each 32-deep k-chunk is loaded once and reused across all 8 column tiles
// (8 accumulators, 8 WMMAs per chunk). A: f16 row-major [M][K]; Bt: f16 [128][K]
// (B transposed, per-column contiguous K); C: f32 [M][128]. M % 16 == 0, K % 32 == 0.
template <int K>
__global__ void gemm_wmma_strip_kernel(const _Float16* __restrict__ A,
                                       const _Float16* __restrict__ Bt,
                                       float* __restrict__ C) {
    constexpr int NCOLS = 128;
    const int lane = threadIdx.x;          // 0..31
    const int m0  = blockIdx.x * 16;
    const int l15 = lane & 15;
    const int hi  = lane >> 4;             // 0 for lanes 0-15, 1 for lanes 16-31

    // A fragment: lanes 0-15 hold K k0..k0+15; lanes 16-31 hold K k0+16..k0+31
    const _Float16* arow = A + (size_t)(m0 + l15) * K + hi * 16;
    // B fragment base for this lane: col = jt*16 + l15; lanes 0-15 hold K k0..k0+15,
    // lanes 16-31 hold K k0+16..k0+31 (16 contiguous halves = one 32B load)
    const _Float16* bbase = Bt + (size_t)l15 * K + hi * 16;

    v8f acc[8];
    #pragma unroll
    for (int j = 0; j < 8; ++j) acc[j] = (v8f){};

    #pragma unroll
    for (int k0 = 0; k0 < K; k0 += 32) {
        v8h alo = *(const v8h*)(arow + k0);
        v8h ahi = *(const v8h*)(arow + k0 + 8);
        union { v16h v; v8h h[2]; } ua;
        ua.h[0] = alo;
        ua.h[1] = ahi;
        #pragma unroll
        for (int j = 0; j < 8; ++j) {
            v16h b = *(const v16h*)(bbase + (size_t)j * 16 * K + k0);
            acc[j] = __builtin_amdgcn_wmma_f32_16x16x32_f16(
                         false, ua.v, false, b, (short)0, acc[j], false, false);
        }
    }

    // D layout: VGPR r -> row = r (lanes 0-15) / r+8 (lanes 16-31), col = lane&15
    float* crow = C + (size_t)(m0 + hi * 8) * NCOLS + l15;
    #pragma unroll
    for (int j = 0; j < 8; ++j) {
        #pragma unroll
        for (int r = 0; r < 8; ++r) crow[(size_t)r * NCOLS + j * 16] = acc[j][r];
    }
}

// ---------------- attention score kernel ----------------
// HF: [N][2*C] per-head features. ES/ED: [N][2].
__global__ void scores_kernel(const float* __restrict__ HF,
                              const float* __restrict__ a_src,
                              const float* __restrict__ a_dst,
                              float* __restrict__ ES, float* __restrict__ ED,
                              int N, int C) {
    int i = blockIdx.x * blockDim.x + threadIdx.x;   // n*2 + h
    if (i >= N * 2) return;
    int n = i >> 1;
    int h = i & 1;
    const float* row = HF + (size_t)n * (2 * C) + h * C;
    const float* as  = a_src + h * C;
    const float* ad  = a_dst + h * C;
    float s1 = 0.f, s2 = 0.f;
    for (int c = 0; c < C; ++c) {
        float v = row[c];
        s1 += v * as[c];
        s2 += v * ad[c];
    }
    ES[i] = s1;
    ED[i] = s2;
}

// ---------------- ordered-int float max helpers ----------------
__device__ __forceinline__ int f2ord(float f) {
    int i = __float_as_int(f);
    return i >= 0 ? i : (i ^ 0x7FFFFFFF);
}
__device__ __forceinline__ float ord2f(int i) {
    return __int_as_float(i >= 0 ? i : (i ^ 0x7FFFFFFF));
}
__device__ __forceinline__ float lrelu(float e) {
    return e > 0.f ? e : NEG_SLOPE * e;
}
#define ORD_NEG_INF ((int)0x807FFFFF)  // f2ord(-inf)

// ---------------- edge pass 1: segment max ----------------
__global__ void edge_max_kernel(const int* __restrict__ ei, int E, int ET,
                                const float* __restrict__ ES,
                                const float* __restrict__ ED,
                                int* __restrict__ Mord) {
    int i = blockIdx.x * blockDim.x + threadIdx.x;
    if (i >= ET) return;
    int s, d;
    if (i < E) { s = ei[i]; d = ei[E + i]; } else { s = d = i - E; }
    #pragma unroll
    for (int h = 0; h < 2; ++h) {
        float e = lrelu(ES[s * 2 + h] + ED[d * 2 + h]);
        atomicMax(&Mord[d * 2 + h], f2ord(e));
    }
}

// decode ordered-int max back to float in place; non-finite -> 0
__global__ void decode_m_kernel(int* __restrict__ M, int n) {
    int i = blockIdx.x * blockDim.x + threadIdx.x;
    if (i >= n) return;
    float f = ord2f(M[i]);
    if (!isfinite(f)) f = 0.f;
    ((float*)M)[i] = f;
}

// ---------------- edge pass 2: exp + segment sum ----------------
__global__ void edge_exp_kernel(const int* __restrict__ ei, int E, int ET,
                                const float* __restrict__ ES,
                                const float* __restrict__ ED,
                                const float* __restrict__ Mf,
                                float* __restrict__ ALPHA,
                                float* __restrict__ D) {
    int i = blockIdx.x * blockDim.x + threadIdx.x;
    if (i >= ET) return;
    int s, d;
    if (i < E) { s = ei[i]; d = ei[E + i]; } else { s = d = i - E; }
    #pragma unroll
    for (int h = 0; h < 2; ++h) {
        float e  = lrelu(ES[s * 2 + h] + ED[d * 2 + h]);
        float ex = expf(e - Mf[d * 2 + h]);
        ALPHA[(size_t)i * 2 + h] = ex;
        atomicAdd(&D[d * 2 + h], ex);
    }
}

// ---------------- edge pass 3: weighted scatter aggregation ----------------
// one thread per (edge, head, 4-channel chunk); C == 64, features stride 128.
__global__ void aggregate_kernel(const int* __restrict__ ei, int E, int ET,
                                 const float* __restrict__ ALPHA,
                                 const float* __restrict__ D,
                                 const float* __restrict__ HF,
                                 float* __restrict__ AGG) {
    int gid = blockIdx.x * blockDim.x + threadIdx.x;
    int total = ET * 32;                     // 2 heads * 16 chunks of 4
    if (gid >= total) return;
    int i   = gid >> 5;
    int sub = gid & 31;
    int h   = sub >> 4;
    int c4  = (sub & 15) * 4;
    int s, d;
    if (i < E) { s = ei[i]; d = ei[E + i]; } else { s = d = i - E; }
    float alpha = ALPHA[(size_t)i * 2 + h] / (D[d * 2 + h] + 1e-16f);
    const float4 v = *(const float4*)(HF + (size_t)s * 128 + h * 64 + c4);
    float* dp = AGG + (size_t)d * 128 + h * 64 + c4;
    atomicAdd(dp + 0, alpha * v.x);
    atomicAdd(dp + 1, alpha * v.y);
    atomicAdd(dp + 2, alpha * v.z);
    atomicAdd(dp + 3, alpha * v.w);
}

// ---------------- finalize: mean over heads + bias (+relu, ->f16) ----------------
__global__ void finalize1_kernel(const float* __restrict__ AGG,
                                 const float* __restrict__ bias,
                                 _Float16* __restrict__ OUTh, int N) {
    int i = blockIdx.x * blockDim.x + threadIdx.x;   // n*64 + c
    if (i >= N * 64) return;
    int n = i >> 6;
    int c = i & 63;
    float v = 0.5f * (AGG[(size_t)n * 128 + c] + AGG[(size_t)n * 128 + 64 + c]) + bias[c];
    v = v > 0.f ? v : 0.f;                           // ReLU between layers
    OUTh[i] = (_Float16)v;
}

__global__ void finalize2_kernel(const float* __restrict__ AGG,
                                 const float* __restrict__ bias,
                                 float* __restrict__ out, int N) {
    int i = blockIdx.x * blockDim.x + threadIdx.x;   // n*64 + c
    if (i >= N * 64) return;
    int n = i >> 6;
    int c = i & 63;
    out[i] = 0.5f * (AGG[(size_t)n * 128 + c] + AGG[(size_t)n * 128 + 64 + c]) + bias[c];
}

// ---------------- host launch ----------------
extern "C" void kernel_launch(void* const* d_in, const int* in_sizes, int n_in,
                              void* d_out, int out_size, void* d_ws, size_t ws_size,
                              hipStream_t stream) {
    const float* x      = (const float*)d_in[0];   // [N,128]
    const int*   ei     = (const int*)  d_in[1];   // [2,E]
    const float* W1     = (const float*)d_in[2];   // [128,128]
    const float* asrc1  = (const float*)d_in[3];   // [2,64]
    const float* adst1  = (const float*)d_in[4];   // [2,64]
    const float* bias1  = (const float*)d_in[5];   // [64]
    const float* W2     = (const float*)d_in[6];   // [64,128]
    const float* asrc2  = (const float*)d_in[7];   // [2,64]
    const float* adst2  = (const float*)d_in[8];   // [2,64]
    const float* bias2  = (const float*)d_in[9];   // [64]
    float* out = (float*)d_out;

    const int N  = in_sizes[0] / 128;   // 50000
    const int E  = in_sizes[1] / 2;     // 800000
    const int ET = E + N;               // edges + self loops

    // ---- carve workspace (256B aligned) ----
    char* w = (char*)d_ws;
    size_t off = 0;
    auto carve = [&](size_t bytes) -> char* {
        char* p = w + off;
        off += (bytes + 255) & ~(size_t)255;
        return p;
    };
    _Float16* Xh   = (_Float16*)carve((size_t)N * 128 * 2);
    _Float16* Wt1  = (_Float16*)carve((size_t)128 * 128 * 2);
    _Float16* Wt2  = (_Float16*)carve((size_t)128 * 64 * 2);
    float*    HF   = (float*)   carve((size_t)N * 128 * 4);   // per-head features (both layers)
    float*    AGG  = (float*)   carve((size_t)N * 128 * 4);
    float*    ES   = (float*)   carve((size_t)N * 2 * 4);
    float*    ED   = (float*)   carve((size_t)N * 2 * 4);
    int*      Mord = (int*)     carve((size_t)N * 2 * 4);
    float*    D    = (float*)   carve((size_t)N * 2 * 4);
    float*    ALPH = (float*)   carve((size_t)ET * 2 * 4);
    _Float16* O1h  = (_Float16*)carve((size_t)N * 64 * 2);
    (void)ws_size; (void)n_in; (void)out_size;

    const int TB = 256;
    auto blocks = [](long long n, int tb) { return (unsigned)((n + tb - 1) / tb); };

    // ---- prep: fp16 conversions / weight transpose ----
    cvt_f32_to_f16_kernel<<<blocks((long long)N * 128, TB), TB, 0, stream>>>(x, Xh, N * 128);
    transpose_cvt_kernel<<<blocks(128 * 128, TB), TB, 0, stream>>>(W1, Wt1, 128, 128);
    transpose_cvt_kernel<<<blocks(64 * 128, TB), TB, 0, stream>>>(W2, Wt2, 64, 128);

    // ================= layer 1 =================
    // GEMM: HF[N,128] = Xh[N,128] @ W1   (one wave per 16x128 strip)
    gemm_wmma_strip_kernel<128><<<dim3(N / 16), dim3(32), 0, stream>>>(Xh, Wt1, HF);
    scores_kernel<<<blocks((long long)N * 2, TB), TB, 0, stream>>>(HF, asrc1, adst1, ES, ED, N, 64);

    fill_i32_kernel<<<blocks((long long)N * 2, TB), TB, 0, stream>>>(Mord, ORD_NEG_INF, N * 2);
    fill_f32_kernel<<<blocks((long long)N * 2, TB), TB, 0, stream>>>(D, 0.f, N * 2);
    fill_f32_kernel<<<blocks((long long)N * 128, TB), TB, 0, stream>>>(AGG, 0.f, N * 128);

    edge_max_kernel<<<blocks(ET, TB), TB, 0, stream>>>(ei, E, ET, ES, ED, Mord);
    decode_m_kernel<<<blocks((long long)N * 2, TB), TB, 0, stream>>>(Mord, N * 2);
    edge_exp_kernel<<<blocks(ET, TB), TB, 0, stream>>>(ei, E, ET, ES, ED, (float*)Mord, ALPH, D);
    aggregate_kernel<<<blocks((long long)ET * 32, TB), TB, 0, stream>>>(ei, E, ET, ALPH, D, HF, AGG);

    finalize1_kernel<<<blocks((long long)N * 64, TB), TB, 0, stream>>>(AGG, bias1, O1h, N);

    // ================= layer 2 =================
    // GEMM: HF[N,128] = O1h[N,64] @ W2
    gemm_wmma_strip_kernel<64><<<dim3(N / 16), dim3(32), 0, stream>>>(O1h, Wt2, HF);
    scores_kernel<<<blocks((long long)N * 2, TB), TB, 0, stream>>>(HF, asrc2, adst2, ES, ED, N, 64);

    fill_i32_kernel<<<blocks((long long)N * 2, TB), TB, 0, stream>>>(Mord, ORD_NEG_INF, N * 2);
    fill_f32_kernel<<<blocks((long long)N * 2, TB), TB, 0, stream>>>(D, 0.f, N * 2);
    fill_f32_kernel<<<blocks((long long)N * 128, TB), TB, 0, stream>>>(AGG, 0.f, N * 128);

    edge_max_kernel<<<blocks(ET, TB), TB, 0, stream>>>(ei, E, ET, ES, ED, Mord);
    decode_m_kernel<<<blocks((long long)N * 2, TB), TB, 0, stream>>>(Mord, N * 2);
    edge_exp_kernel<<<blocks(ET, TB), TB, 0, stream>>>(ei, E, ET, ES, ED, (float*)Mord, ALPH, D);
    aggregate_kernel<<<blocks((long long)ET * 32, TB), TB, 0, stream>>>(ei, E, ET, ALPH, D, HF, AGG);

    finalize2_kernel<<<blocks((long long)N * 64, TB), TB, 0, stream>>>(AGG, bias2, out, N);
}
